// MultiHeadAttentionLayer_41102837022823
// MI455X (gfx1250) — matrix operand order repaired and checked by reading, not compiled, not run on hardware
//
#include <hip/hip_runtime.h>

typedef __attribute__((ext_vector_type(16))) __bf16 v16bf;
typedef __attribute__((ext_vector_type(8)))  __bf16 v8bf;
typedef __attribute__((ext_vector_type(8)))  float  v8f;
typedef __attribute__((ext_vector_type(4)))  int    v4i;

namespace {
constexpr int B = 2, S = 2048, D = 1024, H = 16, DK = 64;
constexpr float SCALE = 12.5f;  // 100 / sqrt(DK)
}

static __device__ inline v8f wmma_bf16(v16bf a, v16bf b, v8f c) {
  // D = A(16x32 bf16) * B(32x16 bf16) + C(16x16 f32)
  return __builtin_amdgcn_wmma_f32_16x16x32_bf16(false, a, false, b, (short)0, c,
                                                 false, false);
}

// ---- 16-byte global -> LDS copy, async (ASYNCcnt) when the gfx1250 builtin
// ---- exists; otherwise synchronous through VGPRs.
static __device__ inline void async_copy16(const void* g, void* l) {
#if __has_builtin(__builtin_amdgcn_global_load_async_to_lds_b128)
  __attribute__((address_space(1))) v4i* gp =
      (__attribute__((address_space(1))) v4i*)(unsigned long long)g;
  __attribute__((address_space(3))) v4i* lp =
      (__attribute__((address_space(3))) v4i*)(unsigned)(unsigned long long)l;
  __builtin_amdgcn_global_load_async_to_lds_b128(gp, lp, 0, 0);
#else
  *(v4i*)l = *(const v4i*)g;
#endif
}

static __device__ inline void wait_async() {
#if __has_builtin(__builtin_amdgcn_global_load_async_to_lds_b128)
#if __has_builtin(__builtin_amdgcn_s_wait_asynccnt)
  __builtin_amdgcn_s_wait_asynccnt(0);
#else
  asm volatile("s_wait_asynccnt 0" ::: "memory");
#endif
#endif
}

static __device__ inline void cvt8(v16bf& d, int base, const float* p) {
  const float4* p4 = reinterpret_cast<const float4*>(p);
  float4 x = p4[0];
  float4 y = p4[1];
  d[base + 0] = (__bf16)x.x; d[base + 1] = (__bf16)x.y;
  d[base + 2] = (__bf16)x.z; d[base + 3] = (__bf16)x.w;
  d[base + 4] = (__bf16)y.x; d[base + 5] = (__bf16)y.y;
  d[base + 6] = (__bf16)y.z; d[base + 7] = (__bf16)y.w;
}

// A-operand fragment (16x32): lane m=(l&15) holds K=[k0+half*8,+8) then K=[k0+16+half*8,+8)
static __device__ inline v16bf load_a_f32(const float* row, int k0, int half) {
  v16bf d;
  cvt8(d, 0, row + k0 + half * 8);
  cvt8(d, 8, row + k0 + 16 + half * 8);
  return d;
}
// B-operand fragment (32x16): lane n=(l&15) holds B^T row n, K=[k0+half*8,+8) then K=[k0+16+half*8,+8)
static __device__ inline v16bf load_b_f32(const float* row, int k0, int half) {
  v16bf d;
  cvt8(d, 0, row + k0 + half * 8);
  cvt8(d, 8, row + k0 + 16 + half * 8);
  return d;
}
static __device__ inline v16bf load_a_bf(const __bf16* row, int k0, int half) {
  v8bf lo = *reinterpret_cast<const v8bf*>(row + k0 + half * 8);
  v8bf hi = *reinterpret_cast<const v8bf*>(row + k0 + 16 + half * 8);
  return __builtin_shufflevector(lo, hi, 0, 1, 2, 3, 4, 5, 6, 7, 8, 9, 10, 11, 12, 13, 14, 15);
}
static __device__ inline v16bf load_b_bf(const __bf16* row, int k0, int half) {
  v8bf lo = *reinterpret_cast<const v8bf*>(row + k0 + half * 8);
  v8bf hi = *reinterpret_cast<const v8bf*>(row + k0 + 16 + half * 8);
  return __builtin_shufflevector(lo, hi, 0, 1, 2, 3, 4, 5, 6, 7, 8, 9, 10, 11, 12, 13, 14, 15);
}

// ---------------------------------------------------------------------------
// GEMM: Y[m,n] = sum_k A[m,k] * W[n,k] + bias[n]   (i.e. A @ W^T + b)
// MODE 0: dst = bf16 [B,H,S,DK]   (Q / K head-major)
// MODE 1: dst = bf16 [B,H,DK,S]   (V transposed within head)
// MODE 2: dst = f32  [M, D]       (output projection -> d_out)
// ---------------------------------------------------------------------------
template <int MODE>
__global__ __launch_bounds__(256) void proj_gemm_kernel(
    const float* __restrict__ A, const float* __restrict__ W,
    const float* __restrict__ bias, void* __restrict__ dstv) {
  const int tid = threadIdx.x;
  const int wv = tid >> 5, l = tid & 31;
  const int half = l >> 4, lm = l & 15;
  const int m0 = blockIdx.y * 32 + (wv >> 2) * 16;
  const int n0 = blockIdx.x * 64 + (wv & 3) * 16;
  const float* arow = A + (size_t)(m0 + lm) * D;
  const float* wrow = W + (size_t)(n0 + lm) * D;

  v8f c = {};
  for (int k0 = 0; k0 < D; k0 += 32) {
    if (k0 + 64 < D) {                         // -> global_prefetch_b8
      __builtin_prefetch(arow + k0 + 64, 0, 3);
      __builtin_prefetch(wrow + k0 + 64, 0, 3);
    }
    v16bf a = load_a_f32(arow, k0, half);
    v16bf b = load_b_f32(wrow, k0, half);
    c = wmma_bf16(a, b, c);
  }

  const int n = n0 + lm;
  const float bn = bias[n];
#pragma unroll
  for (int r = 0; r < 8; ++r) {
    const int m = m0 + r + 8 * half;  // C/D layout: vgpr r -> row r (+8 for upper half)
    const float y = c[r] + bn;
    if (MODE == 2) {
      ((float*)dstv)[(size_t)m * D + n] = y;
    } else {
      const int bb = m >> 11, s = m & (S - 1);     // m = b*S + s, S = 2048
      const int hh = n >> 6, dk = n & (DK - 1);    // n = h*DK + dk, DK = 64
      if (MODE == 0)
        ((__bf16*)dstv)[((((size_t)bb * H + hh) * S) + s) * DK + dk] = (__bf16)y;
      else
        ((__bf16*)dstv)[((((size_t)bb * H + hh) * DK) + dk) * S + s] = (__bf16)y;
    }
  }
}

// ---------------------------------------------------------------------------
// Flash attention: 8 waves/block share async-DMA'd K/V tiles in LDS
// (double-buffered); each wave owns 16 query rows. A second sweep accumulates
// the mean attention probability per key column.
// ---------------------------------------------------------------------------
__global__ __launch_bounds__(256) void attention_kernel(
    const __bf16* __restrict__ qh, const __bf16* __restrict__ kh,
    const __bf16* __restrict__ vt, float* __restrict__ attn,
    float* __restrict__ matp) {
  __shared__ __align__(16) __bf16 kt[2][32][DK];     // 2 x 4KB key tiles
  __shared__ __align__(16) __bf16 vtile[2][DK][32];  // 2 x 4KB V^T tiles
  __shared__ __align__(16) __bf16 stage[8][16][32];  // per-wave P transpose staging

  const int tid = threadIdx.x;
  const int wv = tid >> 5, l = tid & 31;
  const int half = l >> 4, lm = l & 15;
  const int b = blockIdx.z, h = blockIdx.y;
  const int q0 = blockIdx.x * 128 + wv * 16;
  const size_t bh = (size_t)b * H + h;

  const __bf16* qrow = qh + (bh * S + q0 + lm) * DK;
  const __bf16* kbase = kh + bh * S * DK;
  const __bf16* vtbase = vt + bh * DK * S;

  // cooperative loaders: one 16B chunk per thread per tile (256 chunks = 4KB)
  const int krow_ld = tid >> 3, kcol_ld = (tid & 7) * 8;   // 32 rows x 8 chunks
  const int vrow_ld = tid >> 2, vcol_ld = (tid & 3) * 8;   // 64 rows x 4 chunks
  auto load_k = [&](int kk0, int buf) {
    async_copy16(kbase + (size_t)(kk0 + krow_ld) * DK + kcol_ld,
                 &kt[buf][krow_ld][kcol_ld]);
  };
  auto load_v = [&](int kk0, int buf) {
    async_copy16(vtbase + (size_t)vrow_ld * S + kk0 + vcol_ld,
                 &vtile[buf][vrow_ld][vcol_ld]);
  };

  const v16bf aq0 = load_a_bf(qrow, 0, half);   // d = 0..31
  const v16bf aq1 = load_a_bf(qrow, 32, half);  // d = 32..63

  v8f acc0 = {}, acc1 = {}, acc2 = {}, acc3 = {};
  float mrow[8], lrow[8];
#pragma unroll
  for (int r = 0; r < 8; ++r) { mrow[r] = -3.0e38f; lrow[r] = 0.0f; }

  load_k(0, 0);
  load_v(0, 0);
  for (int kk0 = 0; kk0 < S; kk0 += 32) {
    const int buf = (kk0 >> 5) & 1;
    wait_async();
    __syncthreads();                        // tiles[buf] ready, tiles[buf^1] free
    if (kk0 + 32 < S) { load_k(kk0 + 32, buf ^ 1); load_v(kk0 + 32, buf ^ 1); }

    // ---- scores: two 16x16 tiles over key columns [kk0, kk0+32) ----
    const __bf16* kr0 = &kt[buf][lm][0];
    const __bf16* kr1 = &kt[buf][16 + lm][0];
    v8f s0 = {}, s1 = {};
    s0 = wmma_bf16(aq0, load_b_bf(kr0, 0, half), s0);
    s0 = wmma_bf16(aq1, load_b_bf(kr0, 32, half), s0);
    s1 = wmma_bf16(aq0, load_b_bf(kr1, 0, half), s1);
    s1 = wmma_bf16(aq1, load_b_bf(kr1, 32, half), s1);

    // ---- online softmax (rows live across 16 lanes of each half) ----
    float resc[8];
#pragma unroll
    for (int r = 0; r < 8; ++r) {
      s0[r] *= SCALE;
      s1[r] *= SCALE;
      float v = fmaxf(s0[r], s1[r]);
      for (int off = 1; off < 16; off <<= 1) v = fmaxf(v, __shfl_xor(v, off, 32));
      const float mn = fmaxf(mrow[r], v);
      resc[r] = __expf(mrow[r] - mn);
      mrow[r] = mn;
      s0[r] = __expf(s0[r] - mn);
      s1[r] = __expf(s1[r] - mn);
      float rs = s0[r] + s1[r];
      for (int off = 1; off < 16; off <<= 1) rs += __shfl_xor(rs, off, 32);
      lrow[r] = lrow[r] * resc[r] + rs;
      acc0[r] *= resc[r];
      acc1[r] *= resc[r];
      acc2[r] *= resc[r];
      acc3[r] *= resc[r];
    }

    // ---- transpose P through LDS: C-layout -> A-layout ----
#pragma unroll
    for (int r = 0; r < 8; ++r) {
      stage[wv][r + 8 * half][lm] = (__bf16)s0[r];
      stage[wv][r + 8 * half][16 + lm] = (__bf16)s1[r];
    }
    asm volatile("s_wait_dscnt 0" ::: "memory");  // cross-lane LDS visibility in-wave
    const v16bf ap = load_a_bf(&stage[wv][lm][0], 0, half);

    // ---- acc += P * V  (B-operand rows = V^T rows in LDS) ----
    acc0 = wmma_bf16(ap, load_b_bf(&vtile[buf][0 + lm][0], 0, half), acc0);
    acc1 = wmma_bf16(ap, load_b_bf(&vtile[buf][16 + lm][0], 0, half), acc1);
    acc2 = wmma_bf16(ap, load_b_bf(&vtile[buf][32 + lm][0], 0, half), acc2);
    acc3 = wmma_bf16(ap, load_b_bf(&vtile[buf][48 + lm][0], 0, half), acc3);
  }

  // ---- epilogue: normalize rows, write attn in [B,S,D] layout for O-proj ----
  float inv_l[8];
#pragma unroll
  for (int r = 0; r < 8; ++r) inv_l[r] = 1.0f / lrow[r];
#pragma unroll
  for (int r = 0; r < 8; ++r) {
    const int qi = q0 + r + 8 * half;
    float* orow = attn + ((size_t)b * S + qi) * D + h * DK;
    orow[0 + lm] = acc0[r] * inv_l[r];
    orow[16 + lm] = acc1[r] * inv_l[r];
    orow[32 + lm] = acc2[r] * inv_l[r];
    orow[48 + lm] = acc3[r] * inv_l[r];
  }

  // ---- second sweep: mean attention probability over (heads, queries) ----
  const float invHS = 1.0f / (float)(H * S);
  float* mp = matp + (size_t)b * S;
  __syncthreads();          // everyone done with sweep-1 tiles before reuse
  load_k(0, 0);
  for (int kk0 = 0; kk0 < S; kk0 += 32) {
    const int buf = (kk0 >> 5) & 1;
    wait_async();
    __syncthreads();
    if (kk0 + 32 < S) load_k(kk0 + 32, buf ^ 1);
#pragma unroll
    for (int t = 0; t < 2; ++t) {
      const __bf16* kr = &kt[buf][t * 16 + lm][0];
      v8f z = {};
      z = wmma_bf16(aq0, load_b_bf(kr, 0, half), z);
      z = wmma_bf16(aq1, load_b_bf(kr, 32, half), z);
      float cs = 0.0f;
#pragma unroll
      for (int r = 0; r < 8; ++r)
        cs += __expf(z[r] * SCALE - mrow[r]) * inv_l[r];
      cs += __shfl_xor(cs, 16, 32);  // merge the two row-halves of this column
      if (half == 0) atomicAdd(&mp[kk0 + t * 16 + lm], cs * invHS);
    }
  }
}

__global__ void zero_kernel(float* __restrict__ p, int n) {
  const int i = blockIdx.x * blockDim.x + threadIdx.x;
  if (i < n) p[i] = 0.0f;
}

extern "C" void kernel_launch(void* const* d_in, const int* in_sizes, int n_in,
                              void* d_out, int out_size, void* d_ws, size_t ws_size,
                              hipStream_t stream) {
  (void)in_sizes; (void)n_in; (void)out_size; (void)ws_size;
  const float* q  = (const float*)d_in[0];
  const float* k  = (const float*)d_in[1];
  const float* v  = (const float*)d_in[2];
  const float* Wq = (const float*)d_in[3];
  const float* bq = (const float*)d_in[4];
  const float* Wk = (const float*)d_in[5];
  const float* bk = (const float*)d_in[6];
  const float* Wv = (const float*)d_in[7];
  const float* bv = (const float*)d_in[8];
  const float* Wo = (const float*)d_in[9];
  const float* bo = (const float*)d_in[10];

  float* out  = (float*)d_out;
  float* matp = out + (size_t)B * S * D;  // mean_atp tail: [B,S]

  char* ws = (char*)d_ws;
  __bf16* qh = (__bf16*)(ws);                               // 8 MB  [B,H,S,DK]
  __bf16* kh = (__bf16*)(ws + (size_t)8 * 1024 * 1024);     // 8 MB  [B,H,S,DK]
  __bf16* vt = (__bf16*)(ws + (size_t)16 * 1024 * 1024);    // 8 MB  [B,H,DK,S]
  float*  at = (float*)(ws + (size_t)24 * 1024 * 1024);     // 16 MB [B,S,D]

  zero_kernel<<<(B * S + 255) / 256, 256, 0, stream>>>(matp, B * S);

  dim3 gP(D / 64, (B * S) / 32);
  proj_gemm_kernel<0><<<gP, 256, 0, stream>>>(q, Wq, bq, qh);
  proj_gemm_kernel<0><<<gP, 256, 0, stream>>>(k, Wk, bk, kh);
  proj_gemm_kernel<1><<<gP, 256, 0, stream>>>(v, Wv, bv, vt);

  dim3 gA(S / 128, H, B);
  attention_kernel<<<gA, 256, 0, stream>>>(qh, kh, vt, at, matp);

  proj_gemm_kernel<2><<<gP, 256, 0, stream>>>(at, Wo, bo, out);
}
